// MultiHeadAttention_88235808129109
// MI455X (gfx1250) — matrix operand order repaired and checked by reading, not compiled, not run on hardware
//
#include <hip/hip_runtime.h>

// ---------------------------------------------------------------------------
// MHA forward for MI455X (gfx1250): bf16 WMMA GEMMs + fp32 softmax.
// D = A x B + C uses v_wmma_f32_16x16x32_bf16 (wave32, fp32 accumulate).
// bf16 tile staging uses the CDNA5 async copy path (GLOBAL_LOAD_ASYNC_TO_LDS,
// ASYNCcnt) instead of VGPR round-trips.
// ---------------------------------------------------------------------------

typedef __bf16 bf16;
typedef __attribute__((ext_vector_type(16))) __bf16   v16bf;
typedef __attribute__((ext_vector_type(8)))  float    v8f;
typedef __attribute__((ext_vector_type(4)))  unsigned uvec4;
typedef __attribute__((ext_vector_type(4)))  float    fvec4;

union FragU { uvec4 u[2]; v16bf f; };

// Stage a ROWS x 32 tile (k contiguous in memory, leading dim ld) into LDS
// laid out [row][32] bf16, converting fp32 -> bf16 on the fly (VGPR path:
// needed because of the conversion).
template<int ROWS>
__device__ __forceinline__ void stage_rowmajor_f32(bf16* __restrict__ lds,
                                                   const float* __restrict__ src,
                                                   int ld, int tid) {
  constexpr int SEGS = ROWS * 4;            // 8-element k segments
  for (int s = tid; s < SEGS; s += 256) {
    const int row = s >> 2;
    const int kq  = (s & 3) << 3;
    const float* g = src + (size_t)row * ld + kq;
    fvec4 a = *(const fvec4*)g;
    fvec4 b = *(const fvec4*)(g + 4);
    bf16* d = lds + row * 32 + kq;
    d[0] = (bf16)a[0]; d[1] = (bf16)a[1]; d[2] = (bf16)a[2]; d[3] = (bf16)a[3];
    d[4] = (bf16)b[0]; d[5] = (bf16)b[1]; d[6] = (bf16)b[2]; d[7] = (bf16)b[3];
  }
}

// Same tile shape, bf16 source: use the CDNA5 async global->LDS copy engine.
// Each lane moves one 16-byte (8 x bf16) segment; tracked by ASYNCcnt.
template<int ROWS>
__device__ __forceinline__ void stage_rowmajor_async(bf16* __restrict__ lds,
                                                     const bf16* __restrict__ src,
                                                     int ld, int tid) {
  constexpr int SEGS = ROWS * 4;            // 8-element k segments
  for (int s = tid; s < SEGS; s += 256) {
    const int row = s >> 2;
    const int kq  = (s & 3) << 3;
    const unsigned long long gaddr =
        (unsigned long long)(const void*)(src + (size_t)row * ld + kq);
    const unsigned ldsoff = (unsigned)(unsigned long long)(lds + row * 32 + kq);
    asm volatile("global_load_async_to_lds_b128 %0, %1, off"
                 :: "v"(ldsoff), "v"(gaddr) : "memory");
  }
}

// Stage a 32 x COLS bf16 tile from [k][n] memory (leading dim ld) into LDS
// laid out [n][32] (k contiguous): transposing scatter, VGPR path.
template<int COLS>
__device__ __forceinline__ void stage_trans(bf16* __restrict__ lds,
                                            const bf16* __restrict__ src,
                                            int ld, int tid) {
  constexpr int NB   = COLS / 8;
  constexpr int SEGS = 32 * NB;
  for (int s = tid; s < SEGS; s += 256) {
    const int k  = s / NB;
    const int nq = (s % NB) * 8;
    union { uvec4 u; bf16 e[8]; } t;
    t.u = *(const uvec4*)(src + (size_t)k * ld + nq);
#pragma unroll
    for (int j = 0; j < 8; ++j) lds[(nq + j) * 32 + k] = t.e[j];
  }
}

// Generic tiled GEMM:  Y[m][n] = scale * sum_k A[m][k] * B[n][k]  (+ bias[n])
// A: row-major MxK (fp32 or bf16). B: row-major NxK (fp32 or bf16), i.e. the
// "W, used as W^T" layout — or, with B_TRANS, bf16 memory laid out [k][n].
// blockIdx.z selects (batch, head) via z = b*16 + h; per-operand offsets are
// off = b*SB + h*SH (pass 0s for unbatched).
template<int BM, int BN, int WM, int WN,
         bool A_F32, bool B_F32, bool B_TRANS, bool OUT_F32>
__global__ __launch_bounds__(256) void gemm_wmma_kernel(
    const void* __restrict__ Ap, unsigned long long aSB, unsigned long long aSH, int lda,
    const void* __restrict__ Bp, unsigned long long bSB, unsigned long long bSH, int ldb,
    void*       __restrict__ Yp, unsigned long long ySB, unsigned long long ySH, int ldy,
    const float* __restrict__ bias, float scale, int K)
{
  constexpr int TM = BM / (WM * 16);
  constexpr int TN = BN / (WN * 16);
  constexpr bool ASYNC_A = !A_F32;                 // bf16 row-major A
  constexpr bool ASYNC_B = (!B_F32) && (!B_TRANS); // bf16 row-major B
  __shared__ __align__(16) bf16 ldsA[BM * 32];
  __shared__ __align__(16) bf16 ldsB[BN * 32];

  const int tid  = threadIdx.x;
  const int lane = tid & 31;
  const int wid  = tid >> 5;
  const int wm   = wid / WN;
  const int wn   = wid % WN;

  const int zb = (int)(blockIdx.z >> 4);
  const int zh = (int)(blockIdx.z & 15);
  const size_t offA = (size_t)zb * aSB + (size_t)zh * aSH;
  const size_t offB = (size_t)zb * bSB + (size_t)zh * bSH;
  const size_t offY = (size_t)zb * ySB + (size_t)zh * ySH;

  const int row0 = (int)blockIdx.y * BM;
  const int col0 = (int)blockIdx.x * BN;

  v8f acc[TM][TN];
  const v8f vzero = {0.f, 0.f, 0.f, 0.f, 0.f, 0.f, 0.f, 0.f};
#pragma unroll
  for (int i = 0; i < TM; ++i)
#pragma unroll
    for (int j = 0; j < TN; ++j) acc[i][j] = vzero;

  // ISA fragment addressing (wave32):
  //  A 16x32 bf16: lane holds row M=lane&15; K pairs at koff=(lane<16?0:8)
  //                for VGPRs 0-3 and koff+16 for VGPRs 4-7  -> 2x b128.
  //  B 32x16 bf16: lane holds col N=lane&15; K pairs at koff=(lane<16?0:8)
  //                for VGPRs 0-3 and koff+16 for VGPRs 4-7  -> 2x b128.
  const int frow  = lane & 15;
  const int akoff = (lane < 16) ? 0 : 8;
  const int bkoff = (lane < 16) ? 0 : 8;

  for (int k0 = 0; k0 < K; k0 += 32) {
    __syncthreads();   // previous tile fully consumed before restaging
    if constexpr (ASYNC_A) {
      const bf16* s = (const bf16*)Ap + offA + (size_t)row0 * lda + k0;
      stage_rowmajor_async<BM>(ldsA, s, lda, tid);
    } else {
      const float* s = (const float*)Ap + offA + (size_t)row0 * lda + k0;
      stage_rowmajor_f32<BM>(ldsA, s, lda, tid);
    }
    if constexpr (B_TRANS) {
      const bf16* s = (const bf16*)Bp + offB + (size_t)k0 * ldb + col0;
      stage_trans<BN>(ldsB, s, ldb, tid);
    } else if constexpr (ASYNC_B) {
      const bf16* s = (const bf16*)Bp + offB + (size_t)col0 * ldb + k0;
      stage_rowmajor_async<BN>(ldsB, s, ldb, tid);
    } else {
      const float* s = (const float*)Bp + offB + (size_t)col0 * ldb + k0;
      stage_rowmajor_f32<BN>(ldsB, s, ldb, tid);
    }
    if constexpr (ASYNC_A || ASYNC_B) {
      asm volatile("s_wait_asynccnt 0x0" ::: "memory");
    }
    __syncthreads();

    v16bf afr[TM];
#pragma unroll
    for (int tm = 0; tm < TM; ++tm) {
      const bf16* p = ldsA + (size_t)(wm * TM * 16 + tm * 16 + frow) * 32 + akoff;
      FragU c;
      c.u[0] = *(const uvec4*)p;
      c.u[1] = *(const uvec4*)(p + 16);
      afr[tm] = c.f;
    }
#pragma unroll
    for (int tn = 0; tn < TN; ++tn) {
      const bf16* p = ldsB + (size_t)(wn * TN * 16 + tn * 16 + frow) * 32 + bkoff;
      FragU c;
      c.u[0] = *(const uvec4*)p;
      c.u[1] = *(const uvec4*)(p + 16);
      const v16bf bfr = c.f;
#pragma unroll
      for (int tm = 0; tm < TM; ++tm) {
        acc[tm][tn] = __builtin_amdgcn_wmma_f32_16x16x32_bf16(
            false, afr[tm], false, bfr, (short)0, acc[tm][tn], false, false);
      }
    }
  }

  // C/D 16x16 f32 layout: lanes 0-15: N=lane, M=r; lanes 16-31: N=lane-16, M=8+r.
  const int mlo = (lane < 16) ? 0 : 8;
#pragma unroll
  for (int tn = 0; tn < TN; ++tn) {
    const int n = col0 + wn * TN * 16 + tn * 16 + frow;
    const float bv = bias ? bias[n] : 0.0f;
#pragma unroll
    for (int tm = 0; tm < TM; ++tm) {
      const int mbase = row0 + wm * TM * 16 + tm * 16 + mlo;
#pragma unroll
      for (int r = 0; r < 8; ++r) {
        const float v = acc[tm][tn][r] * scale + bv;
        const size_t idx = offY + (size_t)(mbase + r) * ldy + n;
        if constexpr (OUT_F32) ((float*)Yp)[idx] = v;
        else                   ((bf16*)Yp)[idx]  = (bf16)v;
      }
    }
  }
}

// Row-wise softmax, in place. One 256-thread block per row.
__global__ __launch_bounds__(256) void softmax_rows(float* __restrict__ data, int ncols) {
  __shared__ float red[8];
  const int tid = threadIdx.x;
  float* p = data + (size_t)blockIdx.x * ncols;

  float m = -3.402823466e38f;
  for (int j = tid; j < ncols; j += 256) m = fmaxf(m, p[j]);
#pragma unroll
  for (int o = 16; o > 0; o >>= 1) m = fmaxf(m, __shfl_xor(m, o, 32));
  if ((tid & 31) == 0) red[tid >> 5] = m;
  __syncthreads();
  float bm = red[0];
#pragma unroll
  for (int i = 1; i < 8; ++i) bm = fmaxf(bm, red[i]);
  __syncthreads();

  float s = 0.0f;
  for (int j = tid; j < ncols; j += 256) {
    float e = __expf(p[j] - bm);
    p[j] = e;
    s += e;
  }
#pragma unroll
  for (int o = 16; o > 0; o >>= 1) s += __shfl_xor(s, o, 32);
  if ((tid & 31) == 0) red[tid >> 5] = s;
  __syncthreads();
  float bs = 0.0f;
#pragma unroll
  for (int i = 0; i < 8; ++i) bs += red[i];
  const float inv = 1.0f / bs;
  for (int j = tid; j < ncols; j += 256) p[j] *= inv;
}

// ---------------------------------------------------------------------------
// Launcher
// ---------------------------------------------------------------------------
extern "C" void kernel_launch(void* const* d_in, const int* in_sizes, int n_in,
                              void* d_out, int out_size, void* d_ws, size_t ws_size,
                              hipStream_t stream) {
  (void)in_sizes; (void)n_in; (void)out_size; (void)ws_size;

  constexpr int B = 4, S = 2048, D = 1024, H = 16, DK = 64;
  constexpr int M = B * S;                       // 8192 rows
  typedef unsigned long long u64;

  const float* q  = (const float*)d_in[0];
  const float* k  = (const float*)d_in[1];
  const float* v  = (const float*)d_in[2];
  const float* Wq = (const float*)d_in[3];
  const float* bq = (const float*)d_in[4];
  const float* Wk = (const float*)d_in[5];
  const float* bk = (const float*)d_in[6];
  const float* Wv = (const float*)d_in[7];
  const float* bv = (const float*)d_in[8];
  const float* Wo = (const float*)d_in[9];
  const float* bo = (const float*)d_in[10];

  float* out  = (float*)d_out;                   // (B,S,D)
  float* attn = out + (size_t)M * D;             // (B,H,S,S) fp32

  // Workspace: bf16 Qp, Kp, Vp, Ctx — 4 * 8192*1024 * 2B = 64 MB
  bf16* Qp  = (bf16*)d_ws;
  bf16* Kp  = Qp + (size_t)M * D;
  bf16* Vp  = Kp + (size_t)M * D;
  bf16* Ctx = Vp + (size_t)M * D;

  const u64 SD = (u64)S * D;                     // per-batch stride in Q/K/V/Ctx
  const u64 SS = (u64)S * S;                     // per-head stride in attn

  // 1-3) Projections: Xp = X @ W^T + b  (fp32 in, bf16 out)
  {
    dim3 g(D / 128, M / 128, 1);
    gemm_wmma_kernel<128,128,2,4, true,true,false,false><<<g, 256, 0, stream>>>(
        q,  0ull, 0ull, D,  Wq, 0ull, 0ull, D,  Qp, 0ull, 0ull, D, bq, 1.0f, D);
    gemm_wmma_kernel<128,128,2,4, true,true,false,false><<<g, 256, 0, stream>>>(
        k,  0ull, 0ull, D,  Wk, 0ull, 0ull, D,  Kp, 0ull, 0ull, D, bk, 1.0f, D);
    gemm_wmma_kernel<128,128,2,4, true,true,false,false><<<g, 256, 0, stream>>>(
        v,  0ull, 0ull, D,  Wv, 0ull, 0ull, D,  Vp, 0ull, 0ull, D, bv, 1.0f, D);
  }

  // 4) Scores: attn[b,h] = (Qp_h @ Kp_h^T) / sqrt(64)   (bf16 in via async
  //    global->LDS copies, fp32 out)
  {
    dim3 g(S / 128, S / 128, B * H);
    gemm_wmma_kernel<128,128,2,4, false,false,false,true><<<g, 256, 0, stream>>>(
        Qp,   SD, (u64)DK, D,
        Kp,   SD, (u64)DK, D,
        attn, (u64)H * SS, SS, S,
        nullptr, 0.125f, DK);
  }

  // 5) Softmax over rows of attn (in place)
  softmax_rows<<<B * H * S, 256, 0, stream>>>(attn, S);

  // 6) Ctx: Ctx[b,:,h*64:+64] = P[b,h] @ Vp_h  (P fp32 -> bf16 at staging,
  //    Vp staged transposed: memory is [k][d], LDS needs [d][k])
  {
    dim3 g(DK / 64, S / 128, B * H);
    gemm_wmma_kernel<128,64,4,2, true,false,true,false><<<g, 256, 0, stream>>>(
        attn, (u64)H * SS, SS, S,
        Vp,   SD, (u64)DK, D,
        Ctx,  SD, (u64)DK, D,
        nullptr, 1.0f, S);
  }

  // 7) Output projection: out = Ctx @ Wo^T + bo  (bf16 A via async copies,
  //    fp32 W -> bf16 staging, fp32 out)
  {
    dim3 g(D / 128, M / 128, 1);
    gemm_wmma_kernel<128,128,2,4, false,true,false,true><<<g, 256, 0, stream>>>(
        Ctx, 0ull, 0ull, D,  Wo, 0ull, 0ull, D,  out, 0ull, 0ull, D, bo, 1.0f, D);
  }
}
